// RbfNet_87239375716416
// MI455X (gfx1250) — matrix operand order repaired and checked by reading, not compiled, not run on hardware
//
#include <hip/hip_runtime.h>

typedef __attribute__((ext_vector_type(16))) __bf16 v16bf;
typedef __attribute__((ext_vector_type(8)))  __bf16 v8bf;
typedef __attribute__((ext_vector_type(8)))  float  v8f;

static constexpr int DIM  = 512;    // feature dim (GEMM inner dim)
static constexpr int KC   = 2048;   // number of centers
static constexpr int NCLS = 10;     // classes
static constexpr int MT   = 64;     // rows per block
static constexpr int NT   = 128;    // centers per tile
static constexpr int DC   = 32;     // inner-dim chunk = WMMA K
static constexpr int RSTRIDE = 132; // padded radial row stride (bank-conflict break)

__device__ __forceinline__ void bf16_split(float x, __bf16* hi, __bf16* lo) {
  __bf16 h = (__bf16)x;          // RNE float->bf16
  *hi = h;
  *lo = (__bf16)(x - (float)h);  // residual
}

__device__ __forceinline__ v16bf cat8(v8bf a, v8bf b) {
  return __builtin_shufflevector(a, b, 0,1,2,3,4,5,6,7,8,9,10,11,12,13,14,15);
}

__global__ __launch_bounds__(256) void rbf_fused_kernel(
    const float* __restrict__ X,     // [n, DIM]
    const float* __restrict__ Cn,    // [KC, DIM]
    const float* __restrict__ Beta,  // [KC]
    const float* __restrict__ W,     // [NCLS, KC]
    const float* __restrict__ Bv,    // [NCLS]
    float* __restrict__ Out)         // [n, NCLS]
{
  __shared__ alignas(16) __bf16 Ahi[MT * DC];
  __shared__ alignas(16) __bf16 Alo[MT * DC];
  __shared__ alignas(16) __bf16 Bhi[NT * DC];
  __shared__ alignas(16) __bf16 Blo[NT * DC];
  __shared__ float Rad[MT * RSTRIDE];
  __shared__ float Wt[NCLS * NT];
  __shared__ float x2s[MT];
  __shared__ float c2s[NT];
  __shared__ float bts[NT];

  const int tid    = threadIdx.x;
  const int lane   = tid & 31;
  const int wave   = tid >> 5;
  const int mStrip = wave & 3;    // which 16-row strip
  const int nHalf  = wave >> 2;   // which 64-col half of the tile
  const int blockRow = blockIdx.x * MT;

  // ---- per-block row norms ||x||^2 ----
  if (tid < MT) x2s[tid] = 0.0f;
  __syncthreads();
  {
    const int row = tid >> 2;
    const int q   = tid & 3;
    const float* p = X + (size_t)(blockRow + row) * DIM + q * 128;
    float s = 0.0f;
    #pragma unroll 4
    for (int i = 0; i < 128; i += 4) {
      float4 v = *(const float4*)(p + i);
      s += v.x*v.x + v.y*v.y + v.z*v.z + v.w*v.w;
    }
    atomicAdd(&x2s[row], s);   // ds_add_f32
  }

  float oAcc0 = 0.f, oAcc1 = 0.f, oAcc2 = 0.f;
  const int mOut = tid & 63;
  const int cg   = tid >> 6;   // class group: covers {cg, cg+4, cg+8<10}

  for (int tile = 0; tile < KC / NT; ++tile) {
    const int nBase = tile * NT;
    __syncthreads();   // previous tile fully consumed Wt/c2s/Rad
    if (tid < NT) { c2s[tid] = 0.0f; bts[tid] = Beta[nBase + tid]; }
    __syncthreads();
    {
      // center norms ||c||^2 for this tile + stage W tile
      const int cen = tid & 127;
      const int hh  = tid >> 7;
      const float* p = Cn + (size_t)(nBase + cen) * DIM + hh * 256;
      float s = 0.0f;
      #pragma unroll 4
      for (int i = 0; i < 256; i += 4) {
        float4 v = *(const float4*)(p + i);
        s += v.x*v.x + v.y*v.y + v.z*v.z + v.w*v.w;
      }
      atomicAdd(&c2s[cen], s);
      for (int i = tid; i < NCLS * NT; i += 256)
        Wt[i] = W[(i >> 7) * KC + nBase + (i & 127)];
    }

    v8f acc[4] = {};

    for (int dk = 0; dk < DIM / DC; ++dk) {
      const int d0 = dk * DC;
      __syncthreads();                 // prior WMMA reads done before restage
      // stage A: 64x32 f32 -> bf16 hi/lo (512 float4 tasks)
      #pragma unroll
      for (int it = 0; it < 2; ++it) {
        int idx = tid + it * 256;
        int row = idx >> 3;
        int dv  = (idx & 7) << 2;
        float4 v = *(const float4*)(X + (size_t)(blockRow + row) * DIM + d0 + dv);
        int o = row * DC + dv;
        bf16_split(v.x, &Ahi[o + 0], &Alo[o + 0]);
        bf16_split(v.y, &Ahi[o + 1], &Alo[o + 1]);
        bf16_split(v.z, &Ahi[o + 2], &Alo[o + 2]);
        bf16_split(v.w, &Ahi[o + 3], &Alo[o + 3]);
      }
      // stage B: 128x32 f32 -> bf16 hi/lo (1024 float4 tasks)
      #pragma unroll
      for (int it = 0; it < 4; ++it) {
        int idx = tid + it * 256;
        int row = idx >> 3;
        int dv  = (idx & 7) << 2;
        float4 v = *(const float4*)(Cn + (size_t)(nBase + row) * DIM + d0 + dv);
        int o = row * DC + dv;
        bf16_split(v.x, &Bhi[o + 0], &Blo[o + 0]);
        bf16_split(v.y, &Bhi[o + 1], &Blo[o + 1]);
        bf16_split(v.z, &Bhi[o + 2], &Blo[o + 2]);
        bf16_split(v.w, &Bhi[o + 3], &Blo[o + 3]);
      }
      __syncthreads();

      // A fragment: lane holds row m, K chunks {base..base+7, base+16..base+23}
      const int mRow = mStrip * 16 + (lane & 15);
      const int aOff = mRow * DC + ((lane >> 4) << 3);
      v16bf aHi = cat8(*(const v8bf*)&Ahi[aOff], *(const v8bf*)&Ahi[aOff + 16]);
      v16bf aLo = cat8(*(const v8bf*)&Alo[aOff], *(const v8bf*)&Alo[aOff + 16]);

      #pragma unroll
      for (int j = 0; j < 4; ++j) {
        // B fragment: lane holds column n, K chunks {base..base+7, base+16..base+23}
        const int n    = nHalf * 64 + j * 16 + (lane & 15);
        const int bOff = n * DC + ((lane >> 4) << 3);
        v16bf bHi = cat8(*(const v8bf*)&Bhi[bOff], *(const v8bf*)&Bhi[bOff + 16]);
        v16bf bLo = cat8(*(const v8bf*)&Blo[bOff], *(const v8bf*)&Blo[bOff + 16]);
        // bf16x3 emulation of fp32 dot: hi*hi + hi*lo + lo*hi
        acc[j] = __builtin_amdgcn_wmma_f32_16x16x32_bf16(false, aHi, false, bHi, (short)0, acc[j], false, false);
        acc[j] = __builtin_amdgcn_wmma_f32_16x16x32_bf16(false, aHi, false, bLo, (short)0, acc[j], false, false);
        acc[j] = __builtin_amdgcn_wmma_f32_16x16x32_bf16(false, aLo, false, bHi, (short)0, acc[j], false, false);
      }
    }

    // ---- epilogue: xc -> radial, park in LDS ----
    const int half = lane >> 4;
    const int nl   = lane & 15;
    #pragma unroll
    for (int j = 0; j < 4; ++j) {
      const int n = nHalf * 64 + j * 16 + nl;
      const float c2v = c2s[n];
      const float bt  = bts[n];
      #pragma unroll
      for (int r = 0; r < 8; ++r) {
        const int m = mStrip * 16 + half * 8 + r;  // C/D layout: VGPR r = row r (+8 for hi lanes)
        float d2 = x2s[m] + c2v - 2.0f * acc[j][r];
        d2 = d2 > 0.0f ? d2 : 0.0f;
        float dist = __builtin_amdgcn_sqrtf(d2);
        Rad[m * RSTRIDE + n] = __builtin_amdgcn_exp2f(-1.44269504088896f * bt * dist);
      }
    }
    __syncthreads();

    // ---- tiny GEMM: radial tile x W^T, accumulated in registers ----
    float a0 = 0.f, a1 = 0.f, a2 = 0.f;
    for (int n = 0; n < NT; ++n) {
      float rr = Rad[mOut * RSTRIDE + n];
      a0 += rr * Wt[cg * NT + n];
      a1 += rr * Wt[(cg + 4) * NT + n];
      if (cg < 2) a2 += rr * Wt[(cg + 8) * NT + n];
    }
    oAcc0 += a0; oAcc1 += a1; oAcc2 += a2;
  }

  float* orow = Out + (size_t)(blockRow + mOut) * NCLS;
  orow[cg]     = oAcc0 + Bv[cg];
  orow[cg + 4] = oAcc1 + Bv[cg + 4];
  if (cg < 2) orow[cg + 8] = oAcc2 + Bv[cg + 8];
}

extern "C" void kernel_launch(void* const* d_in, const int* in_sizes, int n_in,
                              void* d_out, int out_size, void* d_ws, size_t ws_size,
                              hipStream_t stream) {
  const float* X    = (const float*)d_in[0];
  const float* Cn   = (const float*)d_in[1];
  const float* Beta = (const float*)d_in[2];
  const float* W    = (const float*)d_in[3];
  const float* Bv   = (const float*)d_in[4];
  float* Out = (float*)d_out;
  const int n = in_sizes[0] / DIM;          // 16384
  dim3 grid(n / MT), block(256);
  rbf_fused_kernel<<<grid, block, 0, stream>>>(X, Cn, Beta, W, Bv, Out);
}
